// SSIMLoss_69148973465689
// MI455X (gfx1250) — matrix operand order repaired and checked by reading, not compiled, not run on hardware
//
#include <hip/hip_runtime.h>

// ---------------------------------------------------------------------------
// SSIM (skimage-style, 7x7 valid windows) for B=64, 512x512 f32 images.
// Memory-bound (128 MiB inputs @ 23.3 TB/s ~= 5.5us): two streaming passes.
// Horizontal 7-tap box sums run on the matrix unit:
//   D = A(16x32 f16 colsums) x Band(32x16 f16 0/1)  via v_wmma_f32_16x16x32_f16
// A-operands use a hi/lo two-term f16 split (v = hi + lo), so each product
// against the exact 0/1 band is exact and the f32 accumulator recovers
// full f32 precision for the window sums (2 WMMAs per channel).
// Vertical 7-tap sums are f32 running sums in LDS (per-column, per-thread).
// ---------------------------------------------------------------------------

typedef __attribute__((ext_vector_type(16))) _Float16 v16h;
typedef __attribute__((ext_vector_type(8)))  float    v8f;

#define BATCH   64
#define IMH     512
#define IMW     512
#define OUTD    506            // 512 - 7 + 1
#define TILE_R  128            // output rows per block
#define TILE_C  16             // output cols per block
#define IN_R    (TILE_R + 6)   // 134 input rows needed
#define IN_C    24             // 22 needed (16 + 6 halo), padded to 24
#define COVN    (49.0f / 48.0f)

// ---------------- pass 1: per-block partial min/max ----------------
__global__ void k_minmax(const float* __restrict__ X, const float* __restrict__ Y,
                         float* __restrict__ pmin, float* __restrict__ pmax) {
    const int blk  = blockIdx.x;          // 0..511 : 8 blocks per sample
    const int b    = blk >> 3;
    const int part = blk & 7;
    const int elems = (IMH * IMW) / 8;    // 32768 per array slice
    const size_t base = (size_t)b * (IMH * IMW) + (size_t)part * elems;
    const float4* x4 = (const float4*)(X + base);
    const float4* y4 = (const float4*)(Y + base);
    const int n4 = elems / 4;             // 8192 float4 per array

    float mn = 3.402823466e38f, mx = -3.402823466e38f;
    for (int i = threadIdx.x; i < n4; i += blockDim.x) {
        float4 a = x4[i];
        mn = fminf(mn, fminf(fminf(a.x, a.y), fminf(a.z, a.w)));
        mx = fmaxf(mx, fmaxf(fmaxf(a.x, a.y), fmaxf(a.z, a.w)));
        float4 c = y4[i];
        mn = fminf(mn, fminf(fminf(c.x, c.y), fminf(c.z, c.w)));
        mx = fmaxf(mx, fmaxf(fmaxf(c.x, c.y), fmaxf(c.z, c.w)));
    }
    __shared__ float smn[256], smx[256];
    smn[threadIdx.x] = mn;
    smx[threadIdx.x] = mx;
    __syncthreads();
    for (int s = 128; s > 0; s >>= 1) {
        if (threadIdx.x < s) {
            smn[threadIdx.x] = fminf(smn[threadIdx.x], smn[threadIdx.x + s]);
            smx[threadIdx.x] = fmaxf(smx[threadIdx.x], smx[threadIdx.x + s]);
        }
        __syncthreads();
    }
    if (threadIdx.x == 0) { pmin[blk] = smn[0]; pmax[blk] = smx[0]; }
}

// ---------------- pass 2: combine partials -> data_range per sample --------
__global__ void k_range(const float* __restrict__ pmin, const float* __restrict__ pmax,
                        float* __restrict__ drb) {
    int b = threadIdx.x;
    if (b < BATCH) {
        float mn = 3.402823466e38f, mx = -3.402823466e38f;
        #pragma unroll
        for (int j = 0; j < 8; ++j) {
            mn = fminf(mn, pmin[b * 8 + j]);
            mx = fmaxf(mx, pmax[b * 8 + j]);
        }
        drb[b] = mx - mn;
    }
}

__global__ void k_zero(float* acc) { acc[0] = 0.0f; }

__device__ __forceinline__ float chv(float x, float y, int ch) {
    switch (ch) {
        case 0:  return x;
        case 1:  return y;
        case 2:  return x * x;
        case 3:  return y * y;
        default: return x * y;
    }
}

// ---------------- pass 3: fused 5-channel box filter + SSIM ----------------
__global__ void __launch_bounds__(256)
k_ssim(const float* __restrict__ X, const float* __restrict__ Y,
       const float* __restrict__ drb, float* __restrict__ acc) {
    __shared__ float Xs[IN_R][IN_C];
    __shared__ float Ys[IN_R][IN_C];
    __shared__ float Cs[TILE_R][IN_C];   // vertical 7-row colsums, one channel at a time
    __shared__ float red[256];

    const int tid  = threadIdx.x;
    const int lane = tid & 31;           // wave32
    const int wv   = tid >> 5;           // 8 waves: wave w owns output rows w*16..w*16+15
    const bool hi  = lane >= 16;
    const int nn   = lane & 15;

    const int b   = blockIdx.z;
    const int gr0 = blockIdx.y * TILE_R;
    const int gc0 = blockIdx.x * TILE_C;
    const size_t sbase = (size_t)b * (IMH * IMW);

    // ---- load input tile (clamped at image edge; clamped pixels are masked out) ----
    for (int idx = tid; idx < IN_R * IN_C; idx += 256) {
        int rr = idx / IN_C, cc = idx % IN_C;
        int r = gr0 + rr; if (r > IMH - 1) r = IMH - 1;
        int c = gc0 + cc; if (c > IMW - 1) c = IMW - 1;
        size_t g = sbase + (size_t)r * IMW + c;
        Xs[rr][cc] = X[g];
        Ys[rr][cc] = Y[g];
    }

    // ---- constant band B-fragment: B[k][n] = 1 iff n <= k <= n+6 ----
    // 32x16 f16 B, contiguous k order used here (A matches): lane<16 holds K=0..15 (2/VGPR), lane>=16 holds K=16..31; N = lane&15
    v16h bfrag;
    #pragma unroll
    for (int e = 0; e < 16; ++e) {
        int k = e + (hi ? 16 : 0);
        bfrag[e] = (k >= nn && k <= nn + 6) ? (_Float16)1.0f : (_Float16)0.0f;
    }

    v8f za = {};
    v8f wacc[5];
    #pragma unroll
    for (int ch = 0; ch < 5; ++ch) wacc[ch] = za;

    const int ccol = tid & 31;   // colsum worker: column 0..21 active
    const int rg   = tid >> 5;   // row group (16 rows each)

    #pragma unroll
    for (int ch = 0; ch < 5; ++ch) {
        __syncthreads();   // previous channel's A-fragment LDS reads complete
        if (ccol < 22) {   // vertical running 7-sum down this column
            int r0 = rg * 16;
            float s = 0.0f;
            #pragma unroll
            for (int i = 0; i < 7; ++i)
                s += chv(Xs[r0 + i][ccol], Ys[r0 + i][ccol], ch);
            Cs[r0][ccol] = s;
            for (int r = r0 + 1; r < r0 + 16; ++r) {
                s += chv(Xs[r + 6][ccol], Ys[r + 6][ccol], ch)
                   - chv(Xs[r - 1][ccol], Ys[r - 1][ccol], ch);
                Cs[r][ccol] = s;
            }
        }
        __syncthreads();   // colsums visible

        // ---- A-fragments: A[m][k] = colsum row (wv*16+m), col k (k<22; pad=0) ----
        // hi/lo split: v = hi + lo exactly (band entries are exact 0/1 in f16,
        // so hi*1 and lo*1 accumulate exactly in f32 -> full f32 window sums).
        // 16x32 f16 A, contiguous k order used here (matches B): lane<16: K = 0..15; lane>=16: K = 16..31; M = lane&15
        const int rowA = wv * 16 + nn;
        v16h ahi, alo;
        #pragma unroll
        for (int e = 0; e < 16; ++e) {
            int k = e + (hi ? 16 : 0);
            float v = (k < 22) ? Cs[rowA][k] : 0.0f;
            _Float16 h = (_Float16)v;
            ahi[e] = h;
            alo[e] = (_Float16)(v - (float)h);
        }
        // D[m][n] = sum_{k=n..n+6} colsum[wv*16+m][k]  -> full 7x7 window sum
        wacc[ch] = __builtin_amdgcn_wmma_f32_16x16x32_f16(
            false, ahi, false, bfrag, (short)0, wacc[ch], false, false);
        wacc[ch] = __builtin_amdgcn_wmma_f32_16x16x32_f16(
            false, alo, false, bfrag, (short)0, wacc[ch], false, false);
    }

    // ---- elementwise SSIM on the D fragments ----
    const float dr = drb[b];
    const float C1 = (0.01f * dr) * (0.01f * dr);
    const float C2 = (0.03f * dr) * (0.03f * dr);
    const float inv = 1.0f / 49.0f;

    float local = 0.0f;
    const int outc = gc0 + nn;
    #pragma unroll
    for (int v = 0; v < 8; ++v) {
        // D layout: VGPR v -> row M=v (lanes 0-15) / M=v+8 (lanes 16-31), N = lane&15
        int outr = gr0 + wv * 16 + v + (hi ? 8 : 0);
        if (outr < OUTD && outc < OUTD) {
            float s1 = wacc[0][v], s2 = wacc[1][v];
            float s11 = wacc[2][v], s22 = wacc[3][v], s12 = wacc[4][v];
            float ux = s1 * inv, uy = s2 * inv;
            float vx  = COVN * (s11 * inv - ux * ux);
            float vy  = COVN * (s22 * inv - uy * uy);
            float vxy = COVN * (s12 * inv - ux * uy);
            float A1 = 2.0f * ux * uy + C1, A2 = 2.0f * vxy + C2;
            float B1 = ux * ux + uy * uy + C1, B2 = vx + vy + C2;
            local += (A1 * A2) / (B1 * B2);
        }
    }

    red[tid] = local;
    __syncthreads();
    for (int s = 128; s > 0; s >>= 1) {
        if (tid < s) red[tid] += red[tid + s];
        __syncthreads();
    }
    if (tid == 0) atomicAdd(acc, red[0]);
}

__global__ void k_final(const float* __restrict__ acc, float* __restrict__ out) {
    out[0] = acc[0] * (1.0f / ((float)BATCH * (float)OUTD * (float)OUTD));
}

// ---------------------------------------------------------------------------
extern "C" void kernel_launch(void* const* d_in, const int* in_sizes, int n_in,
                              void* d_out, int out_size, void* d_ws, size_t ws_size,
                              hipStream_t stream) {
    const float* X = (const float*)d_in[0];   // predictions [64,512,512] f32
    const float* Y = (const float*)d_in[1];   // labels      [64,512,512] f32
    float* wsf  = (float*)d_ws;
    float* accp = wsf;            // [1]   running sum of S
    float* pmin = wsf + 1;        // [512] per-block partial mins
    float* pmax = wsf + 513;      // [512] per-block partial maxs
    float* drb  = wsf + 1025;     // [64]  per-sample data ranges

    k_zero<<<1, 1, 0, stream>>>(accp);
    k_minmax<<<512, 256, 0, stream>>>(X, Y, pmin, pmax);
    k_range<<<1, 64, 0, stream>>>(pmin, pmax, drb);

    dim3 grid((OUTD + TILE_C - 1) / TILE_C,   // 32 col tiles
              (OUTD + TILE_R - 1) / TILE_R,   // 4 row tiles
              BATCH);                          // 64 samples
    k_ssim<<<grid, 256, 0, stream>>>(X, Y, drb, accp);

    k_final<<<1, 1, 0, stream>>>(accp, (float*)d_out);
}
